// HACMIL_GA_52055003628169
// MI455X (gfx1250) — matrix operand order repaired and checked by reading, not compiled, not run on hardware
//
#include <hip/hip_runtime.h>
#include <hip/hip_bf16.h>
#include <math.h>

typedef __attribute__((ext_vector_type(16))) __bf16 v16bf;
typedef __attribute__((ext_vector_type(8)))  __bf16 v8bf;
typedef __attribute__((ext_vector_type(4)))  __bf16 v4bf;
typedef __attribute__((ext_vector_type(8)))  float  v8f;

#define N_BAGS 8
#define S_PAT  8192
#define L_FEAT 1024
#define D_HID  128
#define T1_TOK 5
#define T2_TOK 4
#define NCLS   2
#define NM1_K  1024
#define NEGV   (-1e9f)

// d_out flat layout (reference tuple order)
#define OUT_OUTPUTS 0
#define OUT_SLIDE   8
#define OUT_A1      10
#define OUT_A2S     (OUT_A1 + N_BAGS * T1_TOK * S_PAT)     // 327690
#define OUT_AF1     (OUT_A2S + T2_TOK * N_BAGS)            // 327722
#define OUT_RAW     (OUT_AF1 + N_BAGS * L_FEAT)            // 335914

#define MTILE 64      // rows per block in K1

// ---------------------------------------------------------------------------
// K0: transpose + f32->bf16 convert the two big projection weights.
// wvT/wuT laid out [D_HID][L_FEAT] so each output column is contiguous in K.
// ---------------------------------------------------------------------------
__global__ void k0_convert(const float* __restrict__ v1w,
                           const float* __restrict__ u1w,
                           __bf16* __restrict__ wvT,
                           __bf16* __restrict__ wuT) {
    int idx = blockIdx.x * blockDim.x + threadIdx.x;   // 131072 total
    if (idx >= D_HID * L_FEAT) return;
    int d = idx / L_FEAT, k = idx % L_FEAT;
    wvT[idx] = (__bf16)v1w[k * D_HID + d];
    wuT[idx] = (__bf16)u1w[k * D_HID + d];
}

// ---------------------------------------------------------------------------
// K1: fused gated-attention projection via bf16 WMMA.
// Each block: 64 rows of the flattened [65536,1024] feat matrix.
// 16 waves: waves 0..7 -> V projection col-tiles, waves 8..15 -> U.
// Each wave owns 4 M-subtiles (4 x v8f accumulators).
// K loop (step 32): A (64x32) staged straight-line: one b128 global load ->
// packed bf16 convert -> one ds_store_b64 per thread;
// B (2x128x32 bf16) staged with GLOBAL_LOAD_ASYNC_TO_LDS_B128 (ASYNCcnt),
// then 4x v_wmma_f32_16x16x32_bf16 per wave per step.
// Epilogue: tanh*sigmoid gate, then D->T1 projection, store logits.
// ---------------------------------------------------------------------------
__global__ void __launch_bounds__(512)
k1_gemm_gated(const float* __restrict__ feat,
              const __bf16* __restrict__ wvT, const __bf16* __restrict__ wuT,
              const float* __restrict__ v1b, const float* __restrict__ u1b,
              const float* __restrict__ w1w, const float* __restrict__ w1b,
              float* __restrict__ logits) {
    __shared__ __align__(16) __bf16 As[MTILE][32];       // 4 KB
    __shared__ __align__(16) __bf16 Bt[2][D_HID][32];    // 16 KB
    __shared__ float H[2][MTILE][D_HID];                 // 64 KB

    const int tid  = threadIdx.x;
    const int lane = tid & 31;
    const int w    = tid >> 5;        // wave id 0..15
    const int u    = (w >> 3) & 1;    // 0 = V weights, 1 = U weights
    const int nb   = (w & 7) * 16;    // output-column tile base
    const int g    = lane >> 4;       // half-wave group
    const int m    = lane & 15;       // A row within a 16-row subtile
    const int ncol = nb + (lane & 15);
    const long r0  = (long)blockIdx.x * MTILE;

    v8f acc0 = {}, acc1 = {}, acc2 = {}, acc3 = {};

    // per-thread A staging assignment: 4 consecutive f32 -> 4 bf16 (8 B LDS)
    const int arow = tid >> 3;          // 0..63
    const int ak   = (tid & 7) * 4;     // 0,4,...,28
    const float* asrc_base = feat + (r0 + arow) * (long)L_FEAT + ak;

    // per-thread B staging assignment: 16 contiguous bf16 (32 B) per thread
    const int bf  = tid * 16;
    const int buu = bf >> 12;          // 0 = V, 1 = U
    const int brm = bf & 4095;
    const int bd  = brm >> 5;          // output column 0..127
    const int bkk = brm & 31;          // 0 or 16
    const __bf16* bsrc_base = (buu ? wuT : wvT) + (long)bd * L_FEAT + bkk;
    const unsigned lds_b0 = (unsigned)(uintptr_t)(&Bt[buu][bd][bkk]);
    const unsigned lds_b1 = lds_b0 + 16u;

    for (int kc = 0; kc < L_FEAT; kc += 32) {
        // stage A: one b128 load + packed bf16 convert + one b64 LDS store
        {
            float4 f = *(const float4*)(asrc_base + kc);
            v4bf bv = { (__bf16)f.x, (__bf16)f.y, (__bf16)f.z, (__bf16)f.w };
            *(v4bf*)&As[arow][ak] = bv;
            if (kc + 32 < L_FEAT)
                __builtin_prefetch(asrc_base + kc + 32, 0, 3);
        }
        // stage B: async global -> LDS, 2 x 16B per thread (ASYNCcnt path)
        {
            const __bf16* src = bsrc_base + kc;
            asm volatile("global_load_async_to_lds_b128 %0, %1, off"
                         :: "v"(lds_b0), "v"(src) : "memory");
            asm volatile("global_load_async_to_lds_b128 %0, %1, off"
                         :: "v"(lds_b1), "v"(src + 8) : "memory");
        }
        asm volatile("s_wait_asynccnt 0x0" ::: "memory");
        __syncthreads();

        // B 32x16 bf16 lane layout: N = lane&15, K = g*8+(e&7)+(e>=8?16:0)
        v8bf blo = *(const v8bf*)&Bt[u][ncol][g * 8];
        v8bf bhi = *(const v8bf*)&Bt[u][ncol][g * 8 + 16];
        v16bf b = __builtin_shufflevector(blo, bhi, 0,1,2,3,4,5,6,7,8,9,10,11,12,13,14,15);

        // A 16x32 bf16 lane layout: lane m, group g: K = g*8+(e&7)+(e>=8?16:0)
#define DO_MT(MT, ACC)                                                          \
        {                                                                       \
            v8bf alo = *(const v8bf*)&As[(MT)*16 + m][g * 8];                   \
            v8bf ahi = *(const v8bf*)&As[(MT)*16 + m][g * 8 + 16];              \
            v16bf a = __builtin_shufflevector(alo, ahi,                         \
                        0,1,2,3,4,5,6,7,8,9,10,11,12,13,14,15);                 \
            ACC = __builtin_amdgcn_wmma_f32_16x16x32_bf16(false, a, false, b,   \
                                                          (short)0, ACC, false, false); \
        }
        DO_MT(0, acc0)
        DO_MT(1, acc1)
        DO_MT(2, acc2)
        DO_MT(3, acc3)
#undef DO_MT
        __syncthreads();
    }

    // scatter accumulators: C/D layout -> H[u][M][ncol]
#pragma unroll
    for (int r = 0; r < 8; ++r) {
        int M = r + (g << 3);
        H[u][ 0 + M][ncol] = acc0[r];
        H[u][16 + M][ncol] = acc1[r];
        H[u][32 + M][ncol] = acc2[r];
        H[u][48 + M][ncol] = acc3[r];
    }
    __syncthreads();

    // gate: tanh(v + bv) * sigmoid(u + bu)
    for (int i = tid; i < MTILE * D_HID; i += 512) {
        int row = i >> 7, d = i & 127;
        float xv = H[0][row][d] + v1b[d];
        float xu = H[1][row][d] + u1b[d];
        H[0][row][d] = tanhf(xv) * (1.0f / (1.0f + expf(-xu)));
    }
    __syncthreads();

    // D -> T1 projection, store logits [n*T1+t][s]
    if (tid < MTILE * T1_TOK) {
        int row = tid / T1_TOK, t = tid % T1_TOK;
        float s = w1b[t];
        for (int k = 0; k < D_HID; ++k) s += H[0][row][k] * w1w[k * T1_TOK + t];
        long gr = r0 + row;
        int n = (int)(gr >> 13), sp = (int)(gr & 8191);
        logits[((long)(n * T1_TOK + t)) * S_PAT + sp] = s;
    }
}

// ---------------------------------------------------------------------------
// K2: per (n,t) row: radix-select 1024th-largest, mask ~half of the top-1024
// (hash-based stand-in for the reference's PRNG subset), softmax -> A1.
// ---------------------------------------------------------------------------
__device__ __forceinline__ unsigned f2k(float f) {
    unsigned b = __float_as_uint(f);
    return (b & 0x80000000u) ? ~b : (b | 0x80000000u);   // monotone order map
}

__global__ void __launch_bounds__(256)
k2_mask_softmax(const float* __restrict__ logits, float* __restrict__ outA1) {
    __shared__ float vals[S_PAT];          // 32 KB
    __shared__ unsigned hist[256];
    __shared__ unsigned sh_prefix, sh_mask, sh_krem, sh_tie;
    __shared__ float red[256];
    const int tid = threadIdx.x;
    const int rb  = blockIdx.x;            // 0..39 -> (n,t)
    const float* row = logits + (long)rb * S_PAT;

    for (int s = tid; s < S_PAT; s += 256) vals[s] = row[s];
    if (tid == 0) { sh_prefix = 0u; sh_mask = 0u; sh_krem = NM1_K; }
    __syncthreads();

    // 4-pass byte radix select of the NM1-th largest key
    for (int p = 3; p >= 0; --p) {
        hist[tid] = 0u;
        __syncthreads();
        unsigned pref = sh_prefix, msk = sh_mask;
        for (int s = tid; s < S_PAT; s += 256) {
            unsigned key = f2k(vals[s]);
            if ((key & msk) == pref) atomicAdd(&hist[(key >> (p * 8)) & 255u], 1u);
        }
        __syncthreads();
        if (tid == 0) {
            unsigned k = sh_krem, cum = 0;
            for (int b = 255; b >= 0; --b) {
                unsigned c = hist[b];
                if (cum + c >= k) {
                    sh_krem   = k - cum;
                    sh_prefix = pref | ((unsigned)b << (p * 8));
                    sh_mask   = msk  | (0xFFu << (p * 8));
                    break;
                }
                cum += c;
            }
        }
        __syncthreads();
    }
    unsigned thrK = sh_prefix, krem = sh_krem;
    if (tid == 0) sh_tie = 0u;
    __syncthreads();

    for (int s = tid; s < S_PAT; s += 256) {
        unsigned key = f2k(vals[s]);
        bool in = key > thrK;
        if (key == thrK) { unsigned rk = atomicAdd(&sh_tie, 1u); in = (rk < krem); }
        if (in) {
            unsigned h = (unsigned)s * 2654435761u ^ ((unsigned)rb * 0x9E3779B9u + 0x7F4A7C15u);
            h ^= h >> 13; h *= 0xC2B2AE35u; h ^= h >> 16;
            if (h & 1u) vals[s] = NEGV;    // ~NM1*MASK_DROP masked
        }
    }
    __syncthreads();

    // softmax over the row
    float mx = -INFINITY;
    for (int s = tid; s < S_PAT; s += 256) mx = fmaxf(mx, vals[s]);
    red[tid] = mx; __syncthreads();
    for (int off = 128; off > 0; off >>= 1) {
        if (tid < off) red[tid] = fmaxf(red[tid], red[tid + off]);
        __syncthreads();
    }
    mx = red[0]; __syncthreads();
    float sm = 0.f;
    for (int s = tid; s < S_PAT; s += 256) sm += expf(vals[s] - mx);
    red[tid] = sm; __syncthreads();
    for (int off = 128; off > 0; off >>= 1) {
        if (tid < off) red[tid] += red[tid + off];
        __syncthreads();
    }
    float inv = 1.f / red[0];
    for (int s = tid; s < S_PAT; s += 256)
        outA1[(long)rb * S_PAT + s] = expf(vals[s] - mx) * inv;
}

// ---------------------------------------------------------------------------
// K2b: bag_A1[n][s] = mean over T1 of A1 (float4-wide)
// ---------------------------------------------------------------------------
__global__ void k2b_bagA1(const float* __restrict__ A1out, float* __restrict__ bagA1) {
    int idx4 = blockIdx.x * 256 + threadIdx.x;          // 16384 total
    if (idx4 >= N_BAGS * S_PAT / 4) return;
    int idx = idx4 * 4;
    int n = idx >> 13, s = idx & 8191;
    float4 sum = make_float4(0.f, 0.f, 0.f, 0.f);
    for (int t = 0; t < T1_TOK; ++t) {
        float4 v = *(const float4*)&A1out[((long)(n * T1_TOK + t)) * S_PAT + s];
        sum.x += v.x; sum.y += v.y; sum.z += v.z; sum.w += v.w;
    }
    const float c = 1.0f / T1_TOK;
    sum.x *= c; sum.y *= c; sum.z *= c; sum.w *= c;
    *(float4*)&bagA1[idx] = sum;
}

// ---------------------------------------------------------------------------
// K3: afeat1[n][d] = sum_s bag_A1[n][s] * feat[n][s][d]  (2nd feat pass)
// float4-wide: thread owns 4 consecutive d, 8 s-phases reduced via LDS.
// ---------------------------------------------------------------------------
__global__ void __launch_bounds__(256)
k3_afeat1(const float* __restrict__ feat, const float* __restrict__ bagA1,
          float* __restrict__ out) {
    __shared__ float4 red[8][32];
    int n     = blockIdx.x >> 3;
    int dbase = (blockIdx.x & 7) * 128;
    int lane32 = threadIdx.x & 31;
    int phase  = threadIdx.x >> 5;       // 0..7
    int d = dbase + lane32 * 4;
    const float* fb = feat  + (long)n * S_PAT * L_FEAT;
    const float* bb = bagA1 + n * S_PAT;
    float4 acc = make_float4(0.f, 0.f, 0.f, 0.f);
    for (int s = phase; s < S_PAT; s += 8) {
        float wgt = bb[s];
        float4 v = *(const float4*)&fb[(long)s * L_FEAT + d];
        acc.x += wgt * v.x; acc.y += wgt * v.y; acc.z += wgt * v.z; acc.w += wgt * v.w;
    }
    red[phase][lane32] = acc;
    __syncthreads();
    if (phase == 0) {
        float4 t = red[0][lane32];
        for (int p = 1; p < 8; ++p) {
            float4 v = red[p][lane32];
            t.x += v.x; t.y += v.y; t.z += v.z; t.w += v.w;
        }
        *(float4*)&out[OUT_AF1 + n * L_FEAT + d] = t;
    }
}

// ---------------------------------------------------------------------------
// K4: entire block 2 (tiny): gated attn on [8,1024], top-2 mask, softmax,
// afeat2, class heads, slide head.
// ---------------------------------------------------------------------------
__global__ void __launch_bounds__(256)
k4_block2(const float* __restrict__ af1,
          const float* __restrict__ v2w, const float* __restrict__ v2b,
          const float* __restrict__ u2w, const float* __restrict__ u2b,
          const float* __restrict__ w2w, const float* __restrict__ w2b,
          const float* __restrict__ clsw, const float* __restrict__ clsb,
          const float* __restrict__ slw,  const float* __restrict__ slb,
              float* __restrict__ out) {
    __shared__ float h2[N_BAGS][D_HID];
    __shared__ float A2[T2_TOK][N_BAGS];
    __shared__ float A2s[T2_TOK][N_BAGS];
    __shared__ float af2[T2_TOK * L_FEAT];
    __shared__ float bagA[N_BAGS];
    __shared__ float bagf[L_FEAT];
    const int tid = threadIdx.x;

    for (int p = tid; p < N_BAGS * D_HID; p += 256) {
        int n = p >> 7, d = p & 127;
        float sv = v2b[d], su = u2b[d];
        for (int k = 0; k < L_FEAT; ++k) {
            float a = af1[n * L_FEAT + k];
            sv += a * v2w[k * D_HID + d];
            su += a * u2w[k * D_HID + d];
        }
        h2[n][d] = tanhf(sv) * (1.0f / (1.0f + expf(-su)));
    }
    __syncthreads();

    if (tid < T2_TOK * N_BAGS) {
        int t = tid >> 3, n = tid & 7;
        float s = w2b[t];
        for (int d = 0; d < D_HID; ++d) s += h2[n][d] * w2w[d * T2_TOK + t];
        A2[t][n] = s;
    }
    __syncthreads();

    if (tid < T2_TOK) {
        int t = tid;
        int i0 = 0; float v0 = A2[t][0];
        for (int n = 1; n < N_BAGS; ++n) if (A2[t][n] > v0) { v0 = A2[t][n]; i0 = n; }
        int i1 = (i0 == 0) ? 1 : 0; float v1 = A2[t][i1];
        for (int n = 0; n < N_BAGS; ++n)
            if (n != i0 && A2[t][n] > v1) { v1 = A2[t][n]; i1 = n; }
        unsigned h = (unsigned)(t + 1) * 0x9E3779B9u; h ^= h >> 16; h *= 0x85EBCA6Bu; h ^= h >> 13;
        A2[t][(h & 1u) ? i0 : i1] = NEGV;          // mask 1 of top-2
        for (int n = 0; n < N_BAGS; ++n) out[OUT_RAW + t * N_BAGS + n] = A2[t][n];
        float mx = -INFINITY;
        for (int n = 0; n < N_BAGS; ++n) mx = fmaxf(mx, A2[t][n]);
        float sm = 0.f;
        for (int n = 0; n < N_BAGS; ++n) sm += expf(A2[t][n] - mx);
        float inv = 1.f / sm;
        for (int n = 0; n < N_BAGS; ++n) {
            float v = expf(A2[t][n] - mx) * inv;
            A2s[t][n] = v;
            out[OUT_A2S + t * N_BAGS + n] = v;
        }
    }
    __syncthreads();

    for (int i = tid; i < T2_TOK * L_FEAT; i += 256) {
        int t = i >> 10, d = i & 1023;
        float s = 0.f;
        for (int n = 0; n < N_BAGS; ++n) s += A2s[t][n] * af1[n * L_FEAT + d];
        af2[i] = s;
    }
    __syncthreads();

    if (tid < T2_TOK * NCLS) {
        int t = tid >> 1, c = tid & 1;
        float s = clsb[t * NCLS + c];
        for (int d = 0; d < L_FEAT; ++d) s += af2[t * L_FEAT + d] * clsw[(t * NCLS + c) * L_FEAT + d];
        out[OUT_OUTPUTS + tid] = s;
    }
    if (tid < N_BAGS) {
        float s = 0.f;
        for (int t = 0; t < T2_TOK; ++t) s += A2s[t][tid];
        bagA[tid] = s * (1.0f / T2_TOK);
    }
    __syncthreads();
    for (int d = tid; d < L_FEAT; d += 256) {
        float s = 0.f;
        for (int n = 0; n < N_BAGS; ++n) s += bagA[n] * af1[n * L_FEAT + d];
        bagf[d] = s;
    }
    __syncthreads();
    if (tid < NCLS) {
        float s = slb[tid];
        for (int d = 0; d < L_FEAT; ++d) s += bagf[d] * slw[d * NCLS + tid];
        out[OUT_SLIDE + tid] = s;
    }
}

// ---------------------------------------------------------------------------
extern "C" void kernel_launch(void* const* d_in, const int* in_sizes, int n_in,
                              void* d_out, int out_size, void* d_ws, size_t ws_size,
                              hipStream_t stream) {
    (void)in_sizes; (void)n_in; (void)out_size; (void)ws_size;
    const float* feat = (const float*)d_in[0];
    const float* v1w  = (const float*)d_in[1];
    const float* v1b  = (const float*)d_in[2];
    const float* u1w  = (const float*)d_in[3];
    const float* u1b  = (const float*)d_in[4];
    const float* w1w  = (const float*)d_in[5];
    const float* w1b  = (const float*)d_in[6];
    const float* v2w  = (const float*)d_in[7];
    const float* v2b  = (const float*)d_in[8];
    const float* u2w  = (const float*)d_in[9];
    const float* u2b  = (const float*)d_in[10];
    const float* w2w  = (const float*)d_in[11];
    const float* w2b  = (const float*)d_in[12];
    const float* clsw = (const float*)d_in[13];
    const float* clsb = (const float*)d_in[14];
    const float* slw  = (const float*)d_in[15];
    const float* slb  = (const float*)d_in[16];
    float* out = (float*)d_out;
    char* wsb  = (char*)d_ws;

    __bf16* wvT   = (__bf16*)(wsb + 0);        // 256 KB
    __bf16* wuT   = (__bf16*)(wsb + 262144);   // 256 KB
    float* logits = (float*)(wsb + 524288);    // 40*8192*4 = 1.25 MB
    float* bagA1  = (float*)(wsb + 1835008);   // 8*8192*4 = 256 KB

    k0_convert   <<<512, 256, 0, stream>>>(v1w, u1w, wvT, wuT);
    k1_gemm_gated<<<S_PAT * N_BAGS / MTILE, 512, 0, stream>>>(feat, wvT, wuT,
                                                              v1b, u1b, w1w, w1b, logits);
    k2_mask_softmax<<<N_BAGS * T1_TOK, 256, 0, stream>>>(logits, out + OUT_A1);
    k2b_bagA1    <<<64, 256, 0, stream>>>(out + OUT_A1, bagA1);
    k3_afeat1    <<<64, 256, 0, stream>>>(feat, bagA1, out);
    k4_block2    <<<1, 256, 0, stream>>>(out + OUT_AF1, v2w, v2b, u2w, u2b,
                                         w2w, w2b, clsw, clsb, slw, slb, out);
}
